// SECONDBackbone_58832462021311
// MI455X (gfx1250) — matrix operand order repaired and checked by reading, not compiled, not run on hardware
//
#include <hip/hip_runtime.h>
#include <hip/hip_bf16.h>

typedef __attribute__((ext_vector_type(16))) __bf16 bf16x16;
typedef __attribute__((ext_vector_type(8)))  __bf16 bf16x8;
typedef __attribute__((ext_vector_type(8)))  float  f32x8;

#define CDIV(a, b) (((a) + (b) - 1) / (b))

// ---------------------------------------------------------------- utilities

__global__ void k_fill(float* __restrict__ p, float v, long n) {
  long i = (long)blockIdx.x * blockDim.x + threadIdx.x;
  long st = (long)gridDim.x * blockDim.x;
  for (; i < n; i += st) p[i] = v;
}

// ---------------------------------------------------------------- voxelize
// grid: 10 x 128 x 128, vox 0.8, low (-51.2,-51.2,-5.0); mean of (x,y,z,i)
__global__ void k_voxelize(const float* __restrict__ pts, float* __restrict__ sums,
                           float* __restrict__ cnt, int total_pts, int pts_per_batch) {
  int i = blockIdx.x * blockDim.x + threadIdx.x;
  if (i >= total_pts) return;
  int b = i / pts_per_batch;
  const float* p = pts + (size_t)i * 4;
  float px = p[0], py = p[1], pz = p[2], pi = p[3];
  int cx = (int)((px + 51.2f) / 0.8f);
  int cy = (int)((py + 51.2f) / 0.8f);
  int cz = (int)((pz + 5.0f) / 0.8f);
  if (cx < 0 || cy < 0 || cz < 0 || cx >= 128 || cy >= 128 || cz >= 10) return;
  long lin = (((long)b * 10 + cz) * 128 + cy) * 128 + cx;
  atomicAdd(&sums[lin * 4 + 0], px);
  atomicAdd(&sums[lin * 4 + 1], py);
  atomicAdd(&sums[lin * 4 + 2], pz);
  atomicAdd(&sums[lin * 4 + 3], pi);
  atomicAdd(&cnt[lin], 1.0f);
}

// voxel mean + Linear(4,16)+bias (pre-BN), emit mask
__global__ void k_enc1(const float* __restrict__ sums, const float* __restrict__ cnt,
                       const float* __restrict__ w1, const float* __restrict__ b1,
                       float* __restrict__ h16, float* __restrict__ mask, long Nvox) {
  long v = (long)blockIdx.x * blockDim.x + threadIdx.x;
  if (v >= Nvox) return;
  float c = cnt[v];
  float m = (c > 0.0f) ? 1.0f : 0.0f;
  mask[v] = m;
  float inv = (c > 0.0f) ? (1.0f / c) : 0.0f;
  float vf[4];
#pragma unroll
  for (int j = 0; j < 4; ++j) vf[j] = sums[v * 4 + j] * inv;
#pragma unroll
  for (int o = 0; o < 16; ++o) {
    float h = b1[o];
#pragma unroll
    for (int j = 0; j < 4; ++j) h += vf[j] * w1[j * 16 + o];
    h16[v * 16 + o] = h;
  }
}

// Linear(16,32)+bias (pre-BN), one thread per output element
__global__ void k_linear2(const float* __restrict__ h, const float* __restrict__ w2,
                          const float* __restrict__ b2, float* __restrict__ out, long Nvox) {
  long idx = (long)blockIdx.x * blockDim.x + threadIdx.x;
  if (idx >= Nvox * 32) return;
  int o = (int)(idx & 31);
  long v = idx >> 5;
  float acc = b2[o];
  const float* hr = h + v * 16;
#pragma unroll
  for (int j = 0; j < 16; ++j) acc += hr[j] * w2[j * 32 + o];
  out[idx] = acc;
}

// ---------------------------------------------------------------- masked BN
// stats[0..C-1]=sum(x*m), stats[C..2C-1]=sum(x^2*m), stats[2C]=sum(m)
__global__ void k_bn_stats(const float* __restrict__ x, const float* __restrict__ mask,
                           float* __restrict__ stats, long Nvox, int C) {
  int tid = threadIdx.x;
  int c = tid % C;
  int rpb = blockDim.x / C;
  int r = tid / C;
  float s = 0.0f, sq = 0.0f, n = 0.0f;
  for (long v = (long)blockIdx.x * rpb + r; v < Nvox; v += (long)gridDim.x * rpb) {
    float m = mask ? mask[v] : 1.0f;
    float xv = x[v * C + c] * m;
    s += xv;
    sq += xv * xv;  // m in {0,1} -> (x*m)^2 == x^2*m
    n += m;
  }
  atomicAdd(&stats[c], s);
  atomicAdd(&stats[C + c], sq);
  if (c == 0) atomicAdd(&stats[2 * C], n);
}

// y = relu((x-mu)*g*rsqrt(var+eps)+b) * mask ; write bf16 and/or f32
__global__ void k_bn_apply(const float* __restrict__ x, const float* __restrict__ mask,
                           const float* __restrict__ stats, const float* __restrict__ g,
                           const float* __restrict__ bta, __bf16* __restrict__ outb,
                           float* __restrict__ outf, long Nvox, int C) {
  long idx = (long)blockIdx.x * blockDim.x + threadIdx.x;
  if (idx >= Nvox * C) return;
  int c = (int)(idx % C);
  long v = idx / C;
  float n = fmaxf(stats[2 * C], 1.0f);
  float mu = stats[c] / n;
  float var = fmaxf(stats[C + c] / n - mu * mu, 0.0f);
  float sc = g[c] * rsqrtf(var + 1e-5f);
  float y = (x[idx] - mu) * sc + bta[c];
  y = fmaxf(y, 0.0f);
  if (mask) y *= mask[v];
  if (outb) outb[idx] = (__bf16)y;
  if (outf) outf[idx] = y;
}

// 3x3x3 stride-2 max-pool of the mask (pad 1)
__global__ void k_maxpool_mask(const float* __restrict__ mi, float* __restrict__ mo,
                               int B, int Zi, int Yi, int Xi, int Zo, int Yo, int Xo) {
  long idx = (long)blockIdx.x * blockDim.x + threadIdx.x;
  long Nout = (long)B * Zo * Yo * Xo;
  if (idx >= Nout) return;
  long t = idx;
  int ox = (int)(t % Xo); t /= Xo;
  int oy = (int)(t % Yo); t /= Yo;
  int oz = (int)(t % Zo);
  int b = (int)(t / Zo);
  float mx = 0.0f;
  for (int dz = 0; dz < 3; ++dz)
    for (int dy = 0; dy < 3; ++dy)
      for (int dx = 0; dx < 3; ++dx) {
        int iz = 2 * oz - 1 + dz, iy = 2 * oy - 1 + dy, ix = 2 * ox - 1 + dx;
        if (iz >= 0 && iz < Zi && iy >= 0 && iy < Yi && ix >= 0 && ix < Xi)
          mx = fmaxf(mx, mi[(((long)b * Zi + iz) * Yi + iy) * Xi + ix]);
      }
  mo[idx] = mx;
}

// max over Z -> bf16 BEV feature map [B,Y,X,C]
__global__ void k_bev_max(const __bf16* __restrict__ act, __bf16* __restrict__ outb,
                          int B, int Z, int Y, int X, int C) {
  long idx = (long)blockIdx.x * blockDim.x + threadIdx.x;
  long Nout = (long)B * Y * X * C;
  if (idx >= Nout) return;
  long t = idx;
  int c = (int)(t % C); t /= C;
  int x = (int)(t % X); t /= X;
  int y = (int)(t % Y);
  int b = (int)(t / Y);
  float mx = 0.0f;  // activations are post-ReLU*mask, all >= 0
  for (int z = 0; z < Z; ++z)
    mx = fmaxf(mx, (float)act[((((long)b * Z + z) * Y + y) * X + x) * C + c]);
  outb[idx] = (__bf16)mx;
}

// ---------------------------------------------------------------- weight prep
// (3,3,3,Cin,Cout) fp32 -> (tap, Cout, Cin) bf16
__global__ void k_conv_w(const float* __restrict__ k, __bf16* __restrict__ wT,
                         int Cin, int Cout) {
  long idx = (long)blockIdx.x * blockDim.x + threadIdx.x;
  long n = (long)27 * Cin * Cout;
  if (idx >= n) return;
  int co = (int)(idx % Cout);
  int ci = (int)((idx / Cout) % Cin);
  int tap = (int)(idx / ((long)Cout * Cin));
  wT[((long)tap * Cout + co) * Cin + ci] = (__bf16)k[idx];
}

// (K,N) fp32 -> (N,K) bf16
__global__ void k_mat_w(const float* __restrict__ w, __bf16* __restrict__ wT, int K, int N) {
  long idx = (long)blockIdx.x * blockDim.x + threadIdx.x;
  if (idx >= (long)K * N) return;
  int n = (int)(idx % N);
  int kk = (int)(idx / N);
  wT[(long)n * K + kk] = (__bf16)w[idx];
}

// ---------------------------------------------------------------- WMMA conv
// Implicit GEMM: M = 16 output voxels per wave, N = NT*16 out channels per wave
// (A fragment reused across NT accumulators), K = 27*Cin in chunks of 32.
// Weights for each tap are staged in LDS, shared by the 4 waves of the block.
// OOB halo rows load from a zeroed pad buffer (branchless; keeps EXEC full).
// in: bf16 NDHWC, wT: bf16 (tap, Cout, Cin), out: fp32 NDHWC (pre-BN).
// block = (32,4). grid = (Mtiles/4, Cout/(NT*16)). Grid is always exact here.
template <int NT>
__global__ void k_conv3d_wmma(const __bf16* __restrict__ in, const __bf16* __restrict__ wT,
                              const __bf16* __restrict__ zpad, float* __restrict__ out,
                              int B, int Zi, int Yi, int Xi, int Cin, int Zo, int Yo,
                              int Xo, int Cout, int stride) {
  __shared__ __bf16 smem[64 * 128];  // up to NT*16 x Cin bf16 (16 KB of 320 KB WGP LDS)
  const long Nvox = (long)B * Zo * Yo * Xo;
  const long v0 = ((long)blockIdx.x * blockDim.y + threadIdx.y) * 16;
  const int lane = threadIdx.x;
  const int mrow = lane & 15;
  const int hi = lane >> 4;
  const int tid = threadIdx.y * 32 + lane;
  // decompose this lane's A-row output voxel (clamped; grid is exact anyway)
  long vm = v0 + mrow;
  const bool mvalid = vm < Nvox;
  long t = mvalid ? vm : (Nvox - 1);
  int ox = (int)(t % Xo); t /= Xo;
  int oy = (int)(t % Yo); t /= Yo;
  int oz = (int)(t % Zo);
  int b  = (int)(t / Zo);
  const int iz0 = oz * stride - 1, iy0 = oy * stride - 1, ix0 = ox * stride - 1;
  const int n0 = blockIdx.y * NT * 16;  // first output channel of this block
  const int kchunks = Cin >> 5;
  const int nel = NT * 16 * Cin;        // bf16 elements per staged tap slab
  f32x8 acc[NT];
#pragma unroll
  for (int nc = 0; nc < NT; ++nc) acc[nc] = (f32x8){};

  for (int tap = 0; tap < 27; ++tap) {
    const int dz = tap / 9, dy = (tap / 3) % 3, dx = tap % 3;
    const int iz = iz0 + dz, iy = iy0 + dy, ix = ix0 + dx;
    const bool inb = mvalid && iz >= 0 && iz < Zi && iy >= 0 && iy < Yi &&
                     ix >= 0 && ix < Xi;
    // branchless A base: OOB rows read zeros from the pad buffer
    const __bf16* abase =
        inb ? in + ((long)(((long)b * Zi + iz) * Yi + iy) * Xi + ix) * Cin : zpad;
    const long slab = ((long)tap * Cout + n0) * Cin;
    // ---- stage this tap's (NT*16 x Cin) weight slab into LDS ----
    __syncthreads();  // previous tap's readers done
    for (int e = tid * 8; e < nel; e += 128 * 8)
      *(bf16x8*)(smem + e) = *(const bf16x8*)(wT + slab + e);
    if (tap < 26)  // warm L2 for the next tap's slab (global_prefetch_b8)
      __builtin_prefetch((const void*)(wT + slab + (long)Cout * Cin + (long)tid * 8), 0, 1);
    __syncthreads();  // slab visible to all 4 waves
    // ---- K loop ----
    for (int kc = 0; kc < kchunks; ++kc) {
      const int kb = kc << 5;
      // A fragment (16x32): lanes 0-15 K {0-15}, lanes 16-31 K {16-31}
      union { bf16x16 v; bf16x8 h[2]; } ua;
      ua.h[0] = *(const bf16x8*)(abase + kb + (hi ? 16 : 0));
      ua.h[1] = *(const bf16x8*)(abase + kb + (hi ? 24 : 8));
#pragma unroll
      for (int nc = 0; nc < NT; ++nc) {
        // B fragment (32x16) from LDS: lane = column, 16 consecutive K
        const __bf16* bp = smem + ((nc * 16 + mrow) * Cin + kb + (hi ? 16 : 0));
        union { bf16x16 v; bf16x8 h[2]; } ub;
        ub.h[0] = *(const bf16x8*)(bp);
        ub.h[1] = *(const bf16x8*)(bp + 8);
        acc[nc] = __builtin_amdgcn_wmma_f32_16x16x32_bf16(
            false, ua.v, false, ub.v, (short)0, acc[nc], false, false);
      }
    }
  }
  // C/D layout: VGPR r -> row hi*8+r, column = lane&15
#pragma unroll
  for (int nc = 0; nc < NT; ++nc) {
    const int col = n0 + nc * 16 + mrow;
#pragma unroll
    for (int r = 0; r < 8; ++r) {
      long vv = v0 + hi * 8 + r;
      if (vv < Nvox) out[vv * Cout + col] = acc[nc][r];
    }
  }
}

// Plain WMMA GEMM for BEV 1x1 conv: A [M,K] bf16 row-major, wT [N,K] bf16,
// out [M,N] fp32 (pre-BN). block=(32,4), grid=(Mtiles/4, N/16).
// A-row index is clamped instead of guarded: loads are unconditional, EXEC
// stays full, no zero-splat fills in the inner loop (grids here are exact).
__global__ void k_gemm_wmma(const __bf16* __restrict__ A, const __bf16* __restrict__ wT,
                            float* __restrict__ out, long M, int N, int K) {
  const long mtile = (long)blockIdx.x * blockDim.y + threadIdx.y;
  const long v0 = mtile * 16;
  const int lane = threadIdx.x;
  const int mrow = lane & 15;
  const int hi = lane >> 4;
  long row = v0 + mrow;
  row = (row < M) ? row : (M - 1);  // branchless clamp (always in range here)
  const __bf16* abase = A + row * K;
  const int ncol = blockIdx.y * 16 + mrow;
  const __bf16* bbase = wT + (long)ncol * K;
  const int kchunks = K >> 5;
  f32x8 acc = {};
  for (int kc = 0; kc < kchunks; ++kc) {
    const int kb = kc << 5;
    union { bf16x16 v; bf16x8 h[2]; } ua;
    ua.h[0] = *(const bf16x8*)(abase + kb + (hi ? 16 : 0));
    ua.h[1] = *(const bf16x8*)(abase + kb + (hi ? 24 : 8));
    bf16x16 bfrag = *(const bf16x16*)(bbase + kb + (hi ? 16 : 0));
    acc = __builtin_amdgcn_wmma_f32_16x16x32_bf16(
        false, ua.v, false, bfrag, (short)0, acc, false, false);
  }
#pragma unroll
  for (int r = 0; r < 8; ++r) {
    long vv = v0 + hi * 8 + r;
    if (vv < M) out[vv * N + blockIdx.y * 16 + mrow] = acc[r];
  }
}

// ---------------------------------------------------------------- host driver

extern "C" void kernel_launch(void* const* d_in, const int* in_sizes, int n_in,
                              void* d_out, int out_size, void* d_ws, size_t ws_size,
                              hipStream_t stream) {
  const int B = 2;
  const int Z1 = 10, Y1 = 128, X1 = 128;
  const int Z2 = 5,  Y2 = 64,  X2 = 64;
  const int Z3 = 3,  Y3 = 32,  X3 = 32;
  const long N1 = (long)B * Z1 * Y1 * X1;  // 327680
  const long N2 = (long)B * Z2 * Y2 * X2;  //  40960
  const long N3 = (long)B * Z3 * Y3 * X3;  //   6144
  const long M1 = (long)B * Y1 * X1, M2 = (long)B * Y2 * X2, M3 = (long)B * Y3 * X3;

  // ---- workspace layout ----
  char* w = (char*)d_ws;
  auto alloc = [&](size_t bytes) -> void* {
    void* p = (void*)w;
    w += (bytes + 255) & ~(size_t)255;
    return p;
  };
  float*  sums     = (float*)alloc(N1 * 4 * sizeof(float));
  float*  cnt      = (float*)alloc(N1 * sizeof(float));
  float*  mask1    = (float*)alloc(N1 * sizeof(float));
  float*  mask2    = (float*)alloc(N2 * sizeof(float));
  float*  mask3    = (float*)alloc(N3 * sizeof(float));
  float*  conv_tmp = (float*)alloc(N1 * 32 * sizeof(float));   // largest pre-BN buf
  __bf16* act_a    = (__bf16*)alloc(N1 * 32 * sizeof(__bf16));
  void*   shared   = alloc(N1 * 16 * sizeof(float));           // h16 fp32 == act_b size
  float*  h16      = (float*)shared;
  __bf16* act_b    = (__bf16*)shared;                          // reused after encoder
  __bf16* bevA     = (__bf16*)alloc(M1 * 32 * sizeof(__bf16));
  __bf16* wTb      = (__bf16*)alloc((size_t)27 * 128 * 128 * sizeof(__bf16));
  __bf16* zpad     = (__bf16*)alloc(512);                      // zeroed OOB pad row
  float*  stats    = (float*)alloc(4096);
  if ((size_t)(w - (char*)d_ws) > ws_size) return;

  float* out_f = (float*)d_out;
  float* bev1_out = out_f;
  float* bev2_out = bev1_out + M1 * 32;
  float* bev3_out = bev2_out + M2 * 64;

  auto fill = [&](float* p, float v, long n) {
    int blocks = (int)CDIV(n, 1024L * 256);
    k_fill<<<blocks > 0 ? blocks : 1, 256, 0, stream>>>(p, v, n);
  };
  auto bn_pass = [&](const float* x, const float* mask, const float* g, const float* bb,
                     __bf16* outb, float* outf, long Nvox, int C) {
    fill(stats, 0.0f, 2 * C + 1);
    int rpb = 256 / C;
    int gx = (int)CDIV(Nvox, (long)rpb);
    if (gx > 1024) gx = 1024;
    k_bn_stats<<<gx, 256, 0, stream>>>(x, mask, stats, Nvox, C);
    long n = Nvox * C;
    k_bn_apply<<<(int)CDIV(n, 256L), 256, 0, stream>>>(x, mask, stats, g, bb, outb,
                                                       outf, Nvox, C);
  };
  auto conv_block = [&](const __bf16* src, __bf16* dst, const float* kf, const float* g,
                        const float* bb, int Zi, int Yi, int Xi, int Cin, int Zo, int Yo,
                        int Xo, int Cout, int stride, const float* mOut, long NvoxOut) {
    long wel = (long)27 * Cin * Cout;
    k_conv_w<<<(int)CDIV(wel, 256L), 256, 0, stream>>>(kf, wTb, Cin, Cout);
    long tiles = CDIV(NvoxOut, 16L);
    dim3 blk(32, 4);
    if (Cout == 32) {  // NT=2, one N-block covers all 32 channels
      dim3 grid((unsigned)CDIV(tiles, 4L), 1);
      k_conv3d_wmma<2><<<grid, blk, 0, stream>>>(src, wTb, zpad, conv_tmp, B, Zi, Yi,
                                                 Xi, Cin, Zo, Yo, Xo, Cout, stride);
    } else {           // NT=4: 64ch -> grid.y=1, 128ch -> grid.y=2
      dim3 grid((unsigned)CDIV(tiles, 4L), Cout / 64);
      k_conv3d_wmma<4><<<grid, blk, 0, stream>>>(src, wTb, zpad, conv_tmp, B, Zi, Yi,
                                                 Xi, Cin, Zo, Yo, Xo, Cout, stride);
    }
    bn_pass(conv_tmp, mOut, g, bb, dst, nullptr, NvoxOut, Cout);
  };
  auto bev_head = [&](const __bf16* act, int Z, int Y, int X, int C, const float* wmat,
                      const float* g, const float* bb, float* dst, long M) {
    long n = M * C;
    k_bev_max<<<(int)CDIV(n, 256L), 256, 0, stream>>>(act, bevA, B, Z, Y, X, C);
    k_mat_w<<<(int)CDIV((long)C * C, 256L), 256, 0, stream>>>(wmat, wTb, C, C);
    dim3 grid((unsigned)CDIV(CDIV(M, 16L), 4L), C / 16), blk(32, 4);
    k_gemm_wmma<<<grid, blk, 0, stream>>>(bevA, wTb, conv_tmp, M, C, C);
    bn_pass(conv_tmp, nullptr, g, bb, nullptr, dst, M, C);
  };

  // ---- voxelize + mean ----
  fill(sums, 0.0f, N1 * 4);
  fill(cnt, 0.0f, N1);
  fill((float*)zpad, 0.0f, 128);  // 512 B of zeros for OOB A rows
  int total_pts = in_sizes[0] / 4;
  int pts_per_batch = total_pts / B;
  k_voxelize<<<CDIV(total_pts, 256), 256, 0, stream>>>((const float*)d_in[0], sums, cnt,
                                                       total_pts, pts_per_batch);

  // ---- voxel encoder: Linear(4,16)+BN+ReLU, Linear(16,32)+BN+ReLU ----
  k_enc1<<<(int)CDIV(N1, 256L), 256, 0, stream>>>(sums, cnt, (const float*)d_in[1],
                                                  (const float*)d_in[2], h16, mask1, N1);
  bn_pass(h16, mask1, (const float*)d_in[3], (const float*)d_in[4], nullptr, h16, N1, 16);
  k_linear2<<<(int)CDIV(N1 * 32, 256L), 256, 0, stream>>>(h16, (const float*)d_in[5],
                                                          (const float*)d_in[6],
                                                          conv_tmp, N1);
  bn_pass(conv_tmp, mask1, (const float*)d_in[7], (const float*)d_in[8], act_a, nullptr,
          N1, 32);  // act_a now holds the 32ch bf16 input; h16/act_b free

  // ---- stage 1: 5 submanifold 32ch conv blocks ----
  const float* k32 = (const float*)d_in[9];
  const float* g32 = (const float*)d_in[10];
  const float* b32 = (const float*)d_in[11];
  __bf16 *cur = act_a, *oth = act_b;
  for (int i = 0; i < 5; ++i) {
    conv_block(cur, oth, k32 + (long)i * 27 * 32 * 32, g32 + i * 32, b32 + i * 32,
               Z1, Y1, X1, 32, Z1, Y1, X1, 32, 1, mask1, N1);
    __bf16* t = cur; cur = oth; oth = t;
  }
  bev_head(cur, Z1, Y1, X1, 32, (const float*)d_in[24], (const float*)d_in[25],
           (const float*)d_in[26], bev1_out, M1);

  // ---- stage 2: down 32->64 (stride 2) + 4 subm 64ch ----
  k_maxpool_mask<<<(int)CDIV(N2, 256L), 256, 0, stream>>>(mask1, mask2, B, Z1, Y1, X1,
                                                          Z2, Y2, X2);
  conv_block(cur, oth, (const float*)d_in[12], (const float*)d_in[13],
             (const float*)d_in[14], Z1, Y1, X1, 32, Z2, Y2, X2, 64, 2, mask2, N2);
  { __bf16* t = cur; cur = oth; oth = t; }
  const float* k64 = (const float*)d_in[15];
  const float* g64 = (const float*)d_in[16];
  const float* b64 = (const float*)d_in[17];
  for (int i = 0; i < 4; ++i) {
    conv_block(cur, oth, k64 + (long)i * 27 * 64 * 64, g64 + i * 64, b64 + i * 64,
               Z2, Y2, X2, 64, Z2, Y2, X2, 64, 1, mask2, N2);
    __bf16* t = cur; cur = oth; oth = t;
  }
  bev_head(cur, Z2, Y2, X2, 64, (const float*)d_in[27], (const float*)d_in[28],
           (const float*)d_in[29], bev2_out, M2);

  // ---- stage 3: down 64->128 (stride 2) + 4 subm 128ch ----
  k_maxpool_mask<<<(int)CDIV(N3, 256L), 256, 0, stream>>>(mask2, mask3, B, Z2, Y2, X2,
                                                          Z3, Y3, X3);
  conv_block(cur, oth, (const float*)d_in[18], (const float*)d_in[19],
             (const float*)d_in[20], Z2, Y2, X2, 64, Z3, Y3, X3, 128, 2, mask3, N3);
  { __bf16* t = cur; cur = oth; oth = t; }
  const float* k128 = (const float*)d_in[21];
  const float* g128 = (const float*)d_in[22];
  const float* b128 = (const float*)d_in[23];
  for (int i = 0; i < 4; ++i) {
    conv_block(cur, oth, k128 + (long)i * 27 * 128 * 128, g128 + i * 128, b128 + i * 128,
               Z3, Y3, X3, 128, Z3, Y3, X3, 128, 1, mask3, N3);
    __bf16* t = cur; cur = oth; oth = t;
  }
  bev_head(cur, Z3, Y3, X3, 128, (const float*)d_in[30], (const float*)d_in[31],
           (const float*)d_in[32], bev3_out, M3);
}
